// ConvLSTM_60155311948201
// MI455X (gfx1250) — matrix operand order repaired and checked by reading, not compiled, not run on hardware
//
#include <hip/hip_runtime.h>
#include <hip/hip_bf16.h>

// ---------------------------------------------------------------------------
// LSTM (center-tap "ConvLSTM") for MI455X / gfx1250.
//   B=64, S=1024, CIN=256, H=512, K=CIN+H=768, 4H=2048 gate columns.
// Strategy (v2):
//   * 4 batch tiles (M=16) x 16 h-slices (32 h cols -> 128 gate cols) = 64
//     persistent workgroups, 256 threads (8 waves) each.
//   * Per-slice bf16 weight tile (768x128 = 192KB) resident in LDS.
//   * x is pre-swizzled ONCE into bf16 WMMA A-fragment order per (bt, t)
//     (8KB blocks); h is written by each WG directly in fragment order to a
//     global ping-pong buffer (slice hs == k-step 8+hs, so the scatter is a
//     per-element remap already needed for the LSTM update).
//   * Per-step A staging = contiguous 24KB global->LDS copy, done with the
//     Tensor Data Mover (tensor_load_to_lds, 6-arg form + s_wait_tensorcnt)
//     when the builtin exists, else vectorized uint4 copies.
//   * Each wave: one 16x16 N-tile, 24x v_wmma_f32_16x16x32_bf16 (two
//     accumulator chains). Gate tiles -> LDS -> elementwise LSTM update.
//   * Per-step barrier among the 16 slices of a batch tile: agent-scope
//     release counter + acquire spin, double-buffered h.
// ---------------------------------------------------------------------------

typedef __bf16 bf16_t;
typedef __attribute__((ext_vector_type(16))) __bf16 v16bf;
typedef __attribute__((ext_vector_type(8)))  float  v8f;
typedef __attribute__((ext_vector_type(4)))  unsigned int u32x4;
typedef __attribute__((ext_vector_type(8)))  int i32x8;
typedef __attribute__((ext_vector_type(4)))  int i32x4;

#define NBT     4      // batch tiles (16 rows each)
#define NHS     16     // h slices
#define HSL     32     // h cols per slice
#define GCOLS   128    // 4*HSL gate cols per workgroup
#define NKSTEP  24     // 768 / 32
#define NTILE   8      // GCOLS / 16
#define KTOT    768
#define WFRAG   (NTILE * NKSTEP * 32 * 16)   // 98304 bf16 per slice (192KB)
#define AFRAG   (NKSTEP * 32 * 16)           // 12288 bf16 (24KB)
#define XFRAG   4096                         // bf16 per (bt,t): ks 0..7
#define HFRAG_N 8192                         // bf16 per bt: ks 8..23

// d_ws layout (bytes)
#define WS_WPACK 0u            // NHS * WFRAG bf16            = 3,145,728 B
#define WS_XPACK 3145728u      // NBT*1024*XFRAG bf16         = 33,554,432 B
#define WS_HFRAG 36700160u     // 2 * NBT * HFRAG_N bf16      = 131,072 B
#define WS_CNT   36831232u     // 4 x u32 barrier counters

#define OUT_H_OFF 33554432u    // 64*1024*512
#define OUT_C_OFF (OUT_H_OFF + 32768u)

#if __has_builtin(__builtin_amdgcn_tensor_load_to_lds)
#define USE_TDM 1
#else
#define USE_TDM 0
#endif

__device__ __forceinline__ float fsig(float x) {
    return 1.0f / (1.0f + __expf(-x));
}
__device__ __forceinline__ float ftanh(float x) {
    float e = __expf(-2.0f * x);
    return (1.0f - e) / (1.0f + e);
}

// Fragment mapping helpers (16-bit 16x32 A-matrix layout, ISA 7.12.2):
//   fragment element (lane, s) -> (m = lane&15, kk = kk2 + 8*hi) with
//   hi = lane>>4, kk2 = (s&7) + (s&8 ? 16 : 0)
__device__ __forceinline__ int frag_kk(int lane, int s) {
    return (s & 7) + ((s & 8) ? 16 : 0) + 8 * (lane >> 4);
}

// ---------------------------------------------------------------------------
// Pack W[:, :, 1] (f32, [2048][768][3]) into bf16 B-fragment order:
//   wp[hs][nt][ks][lane][s] : n = lane&15 ; k = ks*32 + frag_kk(lane, s)
//   gcol = nt*16 + n ; row = (gcol>>5)*512 + hs*32 + (gcol&31)
// ---------------------------------------------------------------------------
__global__ void lstm_pack_w(const float* __restrict__ W, bf16_t* __restrict__ wp) {
    int idx = blockIdx.x * 256 + threadIdx.x;          // 1,572,864 total
    int s    = idx & 15;
    int lane = (idx >> 4) & 31;
    int ks   = (idx >> 9) % NKSTEP;
    int rest = idx / (16 * 32 * NKSTEP);               // hs*8 + nt
    int nt   = rest & 7;
    int hs   = rest >> 3;

    int n  = lane & 15;
    int k  = ks * 32 + frag_kk(lane, s);
    int gc = nt * 16 + n;
    int row = (gc >> 5) * 512 + hs * 32 + (gc & 31);
    wp[idx] = (bf16_t)W[((size_t)row * KTOT + k) * 3 + 1];   // center tap
}

// ---------------------------------------------------------------------------
// Pack x (f32, [64][1024][256]) into bf16 A-fragment order per (bt, t):
//   xp[bt][t][ks0..7][lane][s], k = ks*32 + frag_kk(lane, s)
// ---------------------------------------------------------------------------
__global__ void lstm_pack_x(const float* __restrict__ x, bf16_t* __restrict__ xp) {
    int idx = blockIdx.x * 256 + threadIdx.x;          // 16,777,216 total
    int s    = idx & 15;
    int lane = (idx >> 4) & 31;
    int ks   = (idx >> 9) & 7;
    int t    = (idx >> 12) & 1023;
    int bt   = idx >> 22;
    int m    = lane & 15;
    int k    = ks * 32 + frag_kk(lane, s);
    xp[idx] = (bf16_t)x[((size_t)(bt * 16 + m) * 1024 + t) * 256 + k];
}

// ---------------------------------------------------------------------------
// Init ping-pong fragment-order h state from h0 and zero barrier counters.
//   hfrag[parity][bt][hs][lane][s] : col = hs*32 + frag_kk(lane, s)
// ---------------------------------------------------------------------------
__global__ void lstm_init(const float* __restrict__ h0,
                          bf16_t* __restrict__ hfrag,
                          unsigned* __restrict__ cnt) {
    int idx = blockIdx.x * 256 + threadIdx.x;          // NBT*HFRAG_N = 32768
    int s    = idx & 15;
    int lane = (idx >> 4) & 31;
    int hs   = (idx >> 9) & 15;
    int bt   = idx >> 13;
    int m    = lane & 15;
    int col  = hs * 32 + frag_kk(lane, s);
    bf16_t v = (bf16_t)h0[(size_t)(bt * 16 + m) * 512 + col];
    hfrag[idx]                     = v;                // parity 0
    hfrag[NBT * HFRAG_N + idx]     = v;                // parity 1
    if (idx < NBT) cnt[idx] = 0u;
}

#if USE_TDM
// Build D# group1 for a 1-D contiguous tile of `n` 8-byte units.
__device__ __forceinline__ i32x8 tdm_group1(unsigned n) {
    i32x8 g;
    g[0] = 3 << 16;                    // data_size = 3 (8 bytes)
    g[1] = (int)((n & 0xffffu) << 16); // tensor_dim0[15:0]  -> bits 63:48
    g[2] = (int)((n >> 16) | (1u << 16)); // dim0[31:16], tensor_dim1 = 1
    g[3] = (int)((n & 0xffffu) << 16); // tile_dim0          -> bits 127:112
    g[4] = 0;                          // tile_dim1 = tile_dim2 = 0 (unused)
    g[5] = (int)n;                     // tensor_dim0_stride[31:0]
    g[6] = 0;
    g[7] = 0;
    return g;
}

__device__ __forceinline__ void tdm_load_1d(const void* gsrc, unsigned lds_off,
                                            unsigned n_units) {
    unsigned long long ga = (unsigned long long)(uintptr_t)gsrc;
    u32x4 g0;
    g0[0] = 1u;                                        // count=1, user mode
    g0[1] = lds_off;                                   // lds_addr
    g0[2] = (unsigned)(ga & 0xffffffffu);              // global_addr[31:0]
    g0[3] = (unsigned)((ga >> 32) & 0x01ffffffu) | 0x80000000u; // addr[56:32] | type=2
    i32x8 g1 = tdm_group1(n_units);
    i32x4 gz4 = {0, 0, 0, 0};
    i32x8 gz8 = {0, 0, 0, 0, 0, 0, 0, 0};
    // 6-arg form (clang-23 / therock-10.0): (g0, g1, g2, g3, g4, cpol)
    // cpol: TH=0, SCOPE=DEV(2) in bits [4:3]
    __builtin_amdgcn_tensor_load_to_lds(g0, g1, gz4, gz4, gz8, 16);
}
#endif

// ---------------------------------------------------------------------------
// Persistent recurrent kernel. grid = (NHS, NBT), block = 256.
// ---------------------------------------------------------------------------
__launch_bounds__(256, 1)
__global__ void lstm_persistent(const float* __restrict__ bias,
                                const float* __restrict__ c0,
                                const bf16_t* __restrict__ wpack,
                                const bf16_t* __restrict__ xpack,
                                bf16_t* __restrict__ hfrag,
                                unsigned* __restrict__ cnt,
                                float* __restrict__ out) {
    __shared__ bf16_t wlds[WFRAG];          // 192 KB resident weight tile
    __shared__ bf16_t alds[AFRAG];          //  24 KB A fragments
    __shared__ float  glds[16 * GCOLS];     //   8 KB gate outputs
    __shared__ float  clds[16 * HSL];       //   2 KB cell state
    __shared__ float  blds[GCOLS];          //  512 B bias slice

    const int tid  = threadIdx.x;
    const int hs   = blockIdx.x;
    const int bt   = blockIdx.y;
    const int lane = tid & 31;
    const int wave = tid >> 5;

    // --- one-time staging: weights (vectorized), bias, initial c ---
    {
        const uint4* src = (const uint4*)(wpack + (size_t)hs * WFRAG);
        uint4* dst = (uint4*)wlds;
        for (int i = tid; i < WFRAG / 8; i += 256) dst[i] = src[i];
    }
    for (int i = tid; i < GCOLS; i += 256) {
        blds[i] = bias[(i >> 5) * 512 + hs * 32 + (i & 31)];
    }
    for (int e = tid; e < 512; e += 256) {
        int m = e >> 5, c = e & 31;
        clds[e] = c0[(size_t)(bt * 16 + m) * 512 + hs * 32 + c];
    }
    __syncthreads();

    unsigned* counter = cnt + bt;
#if USE_TDM
    const unsigned aldsOff =
        (unsigned)(uintptr_t)(__attribute__((address_space(3))) char*)(void*)alds;
#endif

    for (int t = 0; t < 1024; ++t) {
        // --- batch-group barrier: all 16 slices posted step t-1 ---
        if (t > 0) {
            if (tid == 0) {
                unsigned target = (unsigned)(NHS * t);
                while (__hip_atomic_load(counter, __ATOMIC_ACQUIRE,
                                         __HIP_MEMORY_SCOPE_AGENT) < target) {
                    __builtin_amdgcn_s_sleep(1);
                }
            }
            __syncthreads();
        }

        const bf16_t* xsrc = xpack + (size_t)(bt * 1024 + t) * XFRAG;   // 8 KB
        const bf16_t* hsrc = hfrag + (size_t)(t & 1) * (NBT * HFRAG_N)
                                   + (size_t)bt * HFRAG_N;              // 16 KB

        // --- stage A = [x_t | h] : contiguous 24 KB global -> LDS ---
#if USE_TDM
        if (wave == 0 && lane == 0) {
            tdm_load_1d(xsrc, aldsOff, XFRAG / 4);                 // 1024 x 8B
            tdm_load_1d(hsrc, aldsOff + XFRAG * 2, HFRAG_N / 4);   // 2048 x 8B
#if __has_builtin(__builtin_amdgcn_s_wait_tensorcnt)
            __builtin_amdgcn_s_wait_tensorcnt(0);
#else
            asm volatile("s_wait_tensorcnt 0x0" ::: "memory");
#endif
        }
        __syncthreads();
#else
        {
            const uint4* xs = (const uint4*)xsrc;          // 512 uint4
            const uint4* hh = (const uint4*)hsrc;          // 1024 uint4
            uint4* dst = (uint4*)alds;
            for (int i = tid; i < 512; i += 256)  dst[i] = xs[i];
            for (int i = tid; i < 1024; i += 256) dst[512 + i] = hh[i];
        }
        __syncthreads();
#endif

        // --- WMMA: wave w owns N-tile w; 24 k-steps, 2 accumulator chains ---
        {
            v8f acc0 = {}, acc1 = {};
            const v16bf* afr = (const v16bf*)alds;
            const v16bf* bfr = (const v16bf*)(wlds + (size_t)wave * (NKSTEP * 32 * 16));
            for (int ks = 0; ks < NKSTEP; ks += 2) {
                v16bf a0 = afr[ks * 32 + lane];
                v16bf b0 = bfr[ks * 32 + lane];
                acc0 = __builtin_amdgcn_wmma_f32_16x16x32_bf16(
                           false, a0, false, b0, (short)0, acc0, false, false);
                v16bf a1 = afr[(ks + 1) * 32 + lane];
                v16bf b1 = bfr[(ks + 1) * 32 + lane];
                acc1 = __builtin_amdgcn_wmma_f32_16x16x32_bf16(
                           false, a1, false, b1, (short)0, acc1, false, false);
            }
            v8f acc = acc0 + acc1;
            // D layout: VGPR v -> M = v + 8*(lane>=16), N = lane&15
            int n     = lane & 15;
            int mbase = (lane >> 4) * 8;
#pragma unroll
            for (int v = 0; v < 8; ++v)
                glds[(mbase + v) * GCOLS + wave * 16 + n] = acc[v];
        }
        __syncthreads();

        // --- elementwise LSTM update for this 16x32 tile ---
        bf16_t* hdst = hfrag + (size_t)((t + 1) & 1) * (NBT * HFRAG_N)
                             + (size_t)bt * HFRAG_N + (size_t)hs * 512;
        for (int e = tid; e < 512; e += 256) {
            int m = e >> 5, c = e & 31;
            float iv = fsig (glds[m * GCOLS +       c] + blds[      c]);
            float fv = fsig (glds[m * GCOLS +  32 + c] + blds[ 32 + c]);
            float gv = ftanh(glds[m * GCOLS +  64 + c] + blds[ 64 + c]);
            float ov = fsig (glds[m * GCOLS +  96 + c] + blds[ 96 + c]);
            float cn = fv * clds[e] + iv * gv;
            float hn = ov * ftanh(cn);
            clds[e] = cn;
            int b   = bt * 16 + m;
            int col = hs * 32 + c;
            out[((size_t)b * 1024 + t) * 512 + col] = hn;
            // fragment-order h for next step: this slice == k-step 8+hs
            //   hi = (c>>3)&1 ; lane = m + 16*hi ; kk2 = c - 8*hi
            //   s = (kk2&7) + (kk2>=16 ? 8 : 0)
            {
                int hi  = (c >> 3) & 1;
                int ln  = m + 16 * hi;
                int kk2 = c - 8 * hi;
                int s   = (kk2 & 7) + ((kk2 >= 16) ? 8 : 0);
                hdst[ln * 16 + s] = (bf16_t)hn;
            }
            if (t == 1023) {
                out[OUT_H_OFF + (size_t)b * 512 + col] = hn;
                out[OUT_C_OFF + (size_t)b * 512 + col] = cn;
            }
        }
        __threadfence();       // publish h stores device-wide
        __syncthreads();       // all threads done before posting
        if (tid == 0)
            __hip_atomic_fetch_add(counter, 1u, __ATOMIC_RELEASE,
                                   __HIP_MEMORY_SCOPE_AGENT);
    }
}

// ---------------------------------------------------------------------------
extern "C" void kernel_launch(void* const* d_in, const int* in_sizes, int n_in,
                              void* d_out, int out_size, void* d_ws, size_t ws_size,
                              hipStream_t stream) {
    const float* x  = (const float*)d_in[0];   // [64][1024][256]
    const float* W  = (const float*)d_in[1];   // [2048][768][3]
    const float* b  = (const float*)d_in[2];   // [2048]
    const float* h0 = (const float*)d_in[3];   // [64][512]
    const float* c0 = (const float*)d_in[4];   // [64][512]
    float* out = (float*)d_out;                // output | h_final | c_final

    bf16_t*   wpack = (bf16_t*)((char*)d_ws + WS_WPACK);
    bf16_t*   xpack = (bf16_t*)((char*)d_ws + WS_XPACK);
    bf16_t*   hfrag = (bf16_t*)((char*)d_ws + WS_HFRAG);
    unsigned* cnt   = (unsigned*)((char*)d_ws + WS_CNT);

    lstm_pack_w<<<6144, 256, 0, stream>>>(W, wpack);
    lstm_pack_x<<<65536, 256, 0, stream>>>(x, xpack);
    lstm_init<<<128, 256, 0, stream>>>(h0, hfrag, cnt);

    dim3 grid(NHS, NBT);
    lstm_persistent<<<grid, 256, 0, stream>>>(b, c0, wpack, xpack, hfrag, cnt, out);
}
